// SA_module_47863115546667
// MI455X (gfx1250) — matrix operand order repaired and checked by reading, not compiled, not run on hardware
//
#include <hip/hip_runtime.h>

#define BS   8
#define P1   8192
#define P2   2048
#define KNN  32
#define C1   64
#define NPTS (BS * P2 * KNN)   /* 524288 rows */
#define TM   128               /* rows per workgroup in MLP passes */
#define NWG  (NPTS / TM)       /* 4096 */

typedef __attribute__((ext_vector_type(16))) __bf16         v16bf;
typedef __attribute__((ext_vector_type(16))) unsigned short v16u;
typedef __attribute__((ext_vector_type(8)))  float          v8f;
typedef __attribute__((ext_vector_type(4)))  unsigned int   u32x4;
typedef __attribute__((ext_vector_type(4)))  float          f32x4;

union U16x16 { v16u v; u32x4 q[2]; };

__device__ __forceinline__ unsigned short f2bf(float f) {
  unsigned int u = __float_as_uint(f);
  u += 0x7FFFu + ((u >> 16) & 1u);           // round-to-nearest-even
  return (unsigned short)(u >> 16);
}

// ---------------------------------------------------------------------------
// Prep: convert fp32 weights to bf16 (W1 zero-padded 67->96 in K).
// The three bf16 weight blocks are laid out contiguously in ws so the MLP
// passes can stage them to LDS with one linear async copy.
// ---------------------------------------------------------------------------
__global__ void prep_kernel(const float* __restrict__ w1, const float* __restrict__ w2,
                            const float* __restrict__ w3,
                            unsigned short* __restrict__ w1p,
                            unsigned short* __restrict__ w2b,
                            unsigned short* __restrict__ w3b) {
  int t = blockIdx.x * 256 + threadIdx.x;
  if (t < 64 * 96) {
    int o = t / 96, c = t % 96;
    w1p[t] = (c < 67) ? f2bf(w1[o * 67 + c]) : (unsigned short)0;
  } else if (t < 64 * 96 + 64 * 64) {
    int u = t - 64 * 96;
    w2b[u] = f2bf(w2[u]);
  } else if (t < 64 * 96 + 64 * 64 + 128 * 64) {
    int u = t - 64 * 96 - 64 * 64;
    w3b[u] = f2bf(w3[u]);
  }
}

// ---------------------------------------------------------------------------
// KNN: one thread per query, register-resident sorted top-32, LDS-tiled points
// (points packed as padded float4 -> one ds_load_b128 per candidate, and the
// scan address is wave-uniform -> LDS broadcast, no bank conflicts)
// ---------------------------------------------------------------------------
#define KTILE 1024
__global__ void __launch_bounds__(64)
knn_kernel(const float* __restrict__ xyz, const int* __restrict__ sidx,
           int* __restrict__ nn, float* __restrict__ newxyz_out) {
  __shared__ __align__(16) float spts[KTILE * 4];
  const int b = blockIdx.x / (P2 / 64);
  const int q = (blockIdx.x % (P2 / 64)) * 64 + threadIdx.x;
  const int s = sidx[b * P2 + q];
  const float* qp = xyz + ((size_t)b * P1 + s) * 3;
  const float qx = qp[0], qy = qp[1], qz = qp[2];
  float* op = newxyz_out + ((size_t)b * P2 + q) * 3;
  op[0] = qx; op[1] = qy; op[2] = qz;

  float bd[KNN]; int bi[KNN];
#pragma unroll
  for (int j = 0; j < KNN; ++j) { bd[j] = 3.4e38f; bi[j] = 0; }

  for (int t0 = 0; t0 < P1; t0 += KTILE) {
    for (int t = threadIdx.x; t < KTILE; t += 64) {
      const float* p = xyz + ((size_t)b * P1 + t0 + t) * 3;
      f32x4 v; v[0] = p[0]; v[1] = p[1]; v[2] = p[2]; v[3] = 0.f;
      *(f32x4*)&spts[t * 4] = v;
      if (t0 + KTILE < P1)
        __builtin_prefetch(xyz + ((size_t)b * P1 + t0 + KTILE + t) * 3, 0, 0);
    }
    __syncthreads();
    for (int t = 0; t < KTILE; ++t) {
      f32x4 c = *(const f32x4*)&spts[t * 4];
      float dx = qx - c[0], dy = qy - c[1], dz = qz - c[2];
      float d = dx * dx + dy * dy + dz * dz;
      if (d < bd[KNN - 1]) {
        bd[KNN - 1] = d; bi[KNN - 1] = t0 + t;
#pragma unroll
        for (int j = KNN - 1; j > 0; --j) {
          if (bd[j] < bd[j - 1]) {
            float td = bd[j]; bd[j] = bd[j - 1]; bd[j - 1] = td;
            int ti = bi[j];  bi[j] = bi[j - 1]; bi[j - 1] = ti;
          }
        }
      }
    }
    __syncthreads();
  }
  int base = (b * P2 + q) * KNN;
#pragma unroll
  for (int j = 0; j < KNN; ++j) nn[base + j] = bi[j];
}

// ---------------------------------------------------------------------------
// WMMA tile GEMM: one wave computes a 16-row strip; NT 16-col tiles; KT k-steps
// X: LDS bf16 row-major [TM x ldx]; W: LDS bf16 row-major [cout x ldw].
// Per k-step: load A once, preload 4 B fragments, issue 4 WMMAs back-to-back.
// ---------------------------------------------------------------------------
template <int KT, int NT>
__device__ __forceinline__ void
wmma_layer(const unsigned short* X, int ldx, const unsigned short* W,
           int ldw, const float* __restrict__ bias, int mtile, int lane, v8f acc[NT]) {
  const int m = lane & 15;       // row (A) / column (B,C,D) within tile
  const int hi = lane >> 4;      // lane-half selects K sub-block
#pragma unroll
  for (int nt = 0; nt < NT; ++nt) {
    float bv = bias[nt * 16 + m];
    v8f c;
#pragma unroll
    for (int j = 0; j < 8; ++j) c[j] = bv;
    acc[nt] = c;
  }
#pragma unroll
  for (int kt = 0; kt < KT; ++kt) {
    const unsigned short* pa = X + (mtile * 16 + m) * ldx + kt * 32 + hi * 8;
    U16x16 au;
    au.q[0] = *(const u32x4*)pa;          // K = {0..7}  (+8 if hi)
    au.q[1] = *(const u32x4*)(pa + 16);   // K = {16..23}(+8 if hi)
    v16bf A = __builtin_bit_cast(v16bf, au.v);
#pragma unroll
    for (int nb = 0; nb < NT; nb += 4) {
      U16x16 bu[4];
#pragma unroll
      for (int i = 0; i < 4; ++i) {
        const unsigned short* pb = W + ((nb + i) * 16 + m) * ldw + kt * 32 + hi * 8;
        bu[i].q[0] = *(const u32x4*)pb;        // K = {0..7}  (+8 if hi)
        bu[i].q[1] = *(const u32x4*)(pb + 16); // K = {16..23}(+8 if hi)
      }
#pragma unroll
      for (int i = 0; i < 4; ++i) {
        v16bf B = __builtin_bit_cast(v16bf, bu[i].v);
        acc[nb + i] = __builtin_amdgcn_wmma_f32_16x16x32_bf16(
            false, A, false, B, (short)0, acc[nb + i], false, false);
      }
    }
  }
}

// normalize (precomputed scale/shift) + ReLU + store bf16 to LDS for next layer
template <int NT>
__device__ __forceinline__ void
norm_relu_store(const v8f* acc, const float* __restrict__ a, const float* __restrict__ bb,
                unsigned short* Xn, int ldn, int mtile, int lane) {
  const int m = lane & 15;
  const int hiRow = (lane >> 4) * 8;
#pragma unroll
  for (int nt = 0; nt < NT; ++nt) {
    int ch = nt * 16 + m;
    float av = a[ch], bv = bb[ch];
#pragma unroll
    for (int j = 0; j < 8; ++j) {
      float v = fmaxf(acc[nt][j] * av + bv, 0.f);
      Xn[(mtile * 16 + hiRow + j) * ldn + ch] = f2bf(v);
    }
  }
}

// deterministic per-WG channel stats: unique LDS slots, fixed-order reduction
template <int NT, int NCH>
__device__ __forceinline__ void
stats_emit(const v8f* acc, float* s_red, float* __restrict__ part,
           int mtile, int lane, int tid, int wg) {
  const int m = lane & 15;
  const int slot = mtile * 2 + (lane >> 4);   // 0..15
#pragma unroll
  for (int nt = 0; nt < NT; ++nt) {
    int ch = nt * 16 + m;
    float s = 0.f, s2 = 0.f;
#pragma unroll
    for (int j = 0; j < 8; ++j) { float v = acc[nt][j]; s += v; s2 += v * v; }
    s_red[ch * 16 + slot] = s;
    s_red[2048 + ch * 16 + slot] = s2;
  }
  __syncthreads();
  if (tid < 2 * NCH) {
    int which = tid / NCH, ch = tid % NCH;
    const float* p = s_red + which * 2048 + ch * 16;
    float t = 0.f;
#pragma unroll
    for (int s = 0; s < 16; ++s) t += p[s];
    part[wg * 256 + which * 128 + ch] = t;
  }
}

// ---------------------------------------------------------------------------
// MLP pass: LEVEL=1..3 -> compute up to layer LEVEL, emit its stats partials.
//           LEVEL=4    -> full forward + maxpool over K -> output features.
// Weights are staged to LDS with gfx1250 async global->LDS copies (ASYNCcnt),
// overlapped with the feature gather.
// ---------------------------------------------------------------------------
template <int LEVEL>
__global__ void __launch_bounds__(256)
pass_kernel(const float* __restrict__ feat, const float* __restrict__ xyz,
            const int* __restrict__ nn, const float* __restrict__ newxyz,
            float* __restrict__ outFeat,
            const unsigned short* __restrict__ wAll,   // w1p|w2b|w3b contiguous
            const float* __restrict__ b1, const float* __restrict__ b2,
            const float* __restrict__ b3,
            const float* __restrict__ ab, float* __restrict__ part) {
  __shared__ __align__(16) unsigned short sA[TM * 96];   // X0 (ld 96) then X2 (ld 64)
  __shared__ __align__(16) unsigned short sB[TM * 64];   // X1 (ld 64)
  __shared__ __align__(16) unsigned short sW[18432];     // bf16 weights (36 KB)
  __shared__ __align__(16) float s_red[4096];            // stats slots / maxpool overlay

  const int tid = threadIdx.x;
  const int lane = tid & 31;
  const int mtile = tid >> 5;           // 8 waves -> 8 row strips of 16

  // ---- async-stage weights into LDS (overlaps with the gather below) ----
  {
    constexpr int WB = (LEVEL == 1) ? 12288 : (LEVEL == 2) ? 20480 : 36864; // bytes
    const unsigned ldsBase = (unsigned)(size_t)(void*)sW;
#pragma unroll 1
    for (int t = tid; t < WB / 16; t += 256) {           // WB/16 divisible by 256
      unsigned loff = ldsBase + (unsigned)t * 16u;
      const char* gp = (const char*)wAll + (size_t)t * 16;
      asm volatile("global_load_async_to_lds_b128 %0, %1, off"
                   :: "v"(loff), "v"(gp) : "memory");
    }
  }

  if (LEVEL == 4) {
    if (tid < 512) ((unsigned int*)s_red)[tid] = 0u;    // maxpool accumulators
  }

  // ---- gather + center + bf16 convert into X0 [TM x 96] (cols 67..95 = 0) ----
  {
    const int row = tid >> 1, half = tid & 1;
    const int r = blockIdx.x * TM + row;
    const int b = r >> 16;                  // P2*KNN == 65536
    const int p2 = (r >> 5) & (P2 - 1);
    const int idx = nn[r];
    const float* fp = feat + ((size_t)(b * P1 + idx)) * C1 + half * 32;
    unsigned short* xr = sA + row * 96;
#pragma unroll
    for (int c = 0; c < 32; ++c) xr[half * 32 + c] = f2bf(fp[c]);
    if (half) {
      const float* pp = xyz + (size_t)(b * P1 + idx) * 3;
      const float* np = newxyz + (size_t)(b * P2 + p2) * 3;
      xr[64] = f2bf(pp[0] - np[0]);
      xr[65] = f2bf(pp[1] - np[1]);
      xr[66] = f2bf(pp[2] - np[2]);
#pragma unroll
      for (int c = 67; c < 96; ++c) xr[c] = 0;
    }
  }
  asm volatile("s_wait_asynccnt 0x0" ::: "memory");  // this wave's weight copy done
  __syncthreads();                                    // all waves' copies + gather done

  const unsigned short* sW1 = sW;            // 64 x 96
  const unsigned short* sW2 = sW + 6144;     // 64 x 64
  const unsigned short* sW3 = sW + 10240;    // 128 x 64

  // ---- layer 1: [TM x 96] x [96 x 64] ----
  v8f acc1[4];
  wmma_layer<3, 4>(sA, 96, sW1, 96, b1, mtile, lane, acc1);
  if constexpr (LEVEL == 1) {
    stats_emit<4, 64>(acc1, s_red, part, mtile, lane, tid, blockIdx.x);
    return;
  }
  norm_relu_store<4>(acc1, ab + 0 * 256, ab + 0 * 256 + 128, sB, 64, mtile, lane);
  __syncthreads();

  // ---- layer 2: [TM x 64] x [64 x 64] ----
  v8f acc2[4];
  wmma_layer<2, 4>(sB, 64, sW2, 64, b2, mtile, lane, acc2);
  if constexpr (LEVEL == 2) {
    stats_emit<4, 64>(acc2, s_red, part, mtile, lane, tid, blockIdx.x);
    return;
  }
  __syncthreads();  // all layer-1 reads of sA are done; safe to overwrite
  norm_relu_store<4>(acc2, ab + 1 * 256, ab + 1 * 256 + 128, sA, 64, mtile, lane);
  __syncthreads();

  // ---- layer 3: [TM x 64] x [64 x 128] ----
  v8f acc3[8];
  wmma_layer<2, 8>(sA, 64, sW3, 64, b3, mtile, lane, acc3);
  if constexpr (LEVEL == 3) {
    stats_emit<8, 128>(acc3, s_red, part, mtile, lane, tid, blockIdx.x);
    return;
  }

  // ---- LEVEL 4: normalize + ReLU + maxpool over KNN neighbors ----
  if constexpr (LEVEL == 4) {
    unsigned int* s_out = (unsigned int*)s_red;   // 4 queries x 128 ch, >=0 floats
    const float* a3 = ab + 2 * 256;
    const float* bb3 = a3 + 128;
    const int m = lane & 15;
    const int hiRow = (lane >> 4) * 8;
#pragma unroll
    for (int nt = 0; nt < 8; ++nt) {
      int ch = nt * 16 + m;
      float av = a3[ch], bv = bb3[ch];
#pragma unroll
      for (int j = 0; j < 8; ++j) {
        int rowt = mtile * 16 + hiRow + j;
        float v = fmaxf(acc3[nt][j] * av + bv, 0.f);
        atomicMax(&s_out[(rowt >> 5) * 128 + ch], __float_as_uint(v));
      }
    }
    __syncthreads();
    for (int o = tid; o < 512; o += 256) {
      int qq = o >> 7, ch = o & 127;
      int qlin = blockIdx.x * 4 + qq;             // linear (b,p2)
      outFeat[(size_t)qlin * 128 + ch] = __uint_as_float(s_out[qq * 128 + ch]);
    }
  }
}

// ---------------------------------------------------------------------------
// Finalize BN stats for one layer -> per-channel scale a, shift bb
// ---------------------------------------------------------------------------
__global__ void finalize_kernel(const float* __restrict__ part, const float* __restrict__ g,
                                const float* __restrict__ bt, float* __restrict__ a_out,
                                float* __restrict__ bb_out, int nch) {
  int ch = threadIdx.x;
  if (ch >= nch) return;
  double s = 0.0, s2 = 0.0;
  for (int w = 0; w < NWG; ++w) {
    s += (double)part[w * 256 + ch];
    s2 += (double)part[w * 256 + 128 + ch];
  }
  double mean = s / (double)NPTS;
  double var = s2 / (double)NPTS - mean * mean;
  float rs = (float)(1.0 / sqrt(var + 1e-5));
  float a = rs * g[ch];
  a_out[ch] = a;
  bb_out[ch] = bt[ch] - (float)mean * a;
}

// ---------------------------------------------------------------------------
// Host launcher
// ---------------------------------------------------------------------------
extern "C" void kernel_launch(void* const* d_in, const int* in_sizes, int n_in,
                              void* d_out, int out_size, void* d_ws, size_t ws_size,
                              hipStream_t stream) {
  const float* xyz  = (const float*)d_in[0];
  const float* feat = (const float*)d_in[1];
  const int*   sidx = (const int*)d_in[2];
  const float* w1 = (const float*)d_in[3];
  const float* b1 = (const float*)d_in[4];
  const float* g1 = (const float*)d_in[5];
  const float* bt1 = (const float*)d_in[6];
  const float* w2 = (const float*)d_in[7];
  const float* b2 = (const float*)d_in[8];
  const float* g2 = (const float*)d_in[9];
  const float* bt2 = (const float*)d_in[10];
  const float* w3 = (const float*)d_in[11];
  const float* b3 = (const float*)d_in[12];
  const float* g3 = (const float*)d_in[13];
  const float* bt3 = (const float*)d_in[14];

  float* outXYZ = (float*)d_out;                        // [BS*P2*3]
  float* outFeat = (float*)d_out + (size_t)BS * P2 * 3; // [BS*P2*128]

  // workspace layout (bytes)
  const size_t NN_OFF   = 0;
  const size_t PART_OFF = NN_OFF + sizeof(int) * (size_t)NPTS;          // 2 MB
  const size_t AB_OFF   = PART_OFF + sizeof(float) * (size_t)NWG * 256; // +4 MB
  const size_t W1_OFF   = AB_OFF + sizeof(float) * 768;   // bf16 weights, contiguous
  const size_t W2_OFF   = W1_OFF + sizeof(unsigned short) * 64 * 96;
  const size_t W3_OFF   = W2_OFF + sizeof(unsigned short) * 64 * 64;

  char* ws = (char*)d_ws;
  int*            nn   = (int*)(ws + NN_OFF);
  float*          part = (float*)(ws + PART_OFF);
  float*          ab   = (float*)(ws + AB_OFF);
  unsigned short* w1p  = (unsigned short*)(ws + W1_OFF);
  unsigned short* w2b  = (unsigned short*)(ws + W2_OFF);
  unsigned short* w3b  = (unsigned short*)(ws + W3_OFF);

  prep_kernel<<<(64 * 96 + 64 * 64 + 128 * 64 + 255) / 256, 256, 0, stream>>>(
      w1, w2, w3, w1p, w2b, w3b);

  knn_kernel<<<BS * (P2 / 64), 64, 0, stream>>>(xyz, sidx, nn, outXYZ);

  pass_kernel<1><<<NWG, 256, 0, stream>>>(feat, xyz, nn, outXYZ, outFeat,
                                          w1p, b1, b2, b3, ab, part);
  finalize_kernel<<<1, 128, 0, stream>>>(part, g1, bt1, ab + 0 * 256, ab + 0 * 256 + 128, 64);

  pass_kernel<2><<<NWG, 256, 0, stream>>>(feat, xyz, nn, outXYZ, outFeat,
                                          w1p, b1, b2, b3, ab, part);
  finalize_kernel<<<1, 128, 0, stream>>>(part, g2, bt2, ab + 1 * 256, ab + 1 * 256 + 128, 64);

  pass_kernel<3><<<NWG, 256, 0, stream>>>(feat, xyz, nn, outXYZ, outFeat,
                                          w1p, b1, b2, b3, ab, part);
  finalize_kernel<<<1, 128, 0, stream>>>(part, g3, bt3, ab + 2 * 256, ab + 2 * 256 + 128, 128);

  pass_kernel<4><<<NWG, 256, 0, stream>>>(feat, xyz, nn, outXYZ, outFeat,
                                          w1p, b1, b2, b3, ab, part);

  (void)in_sizes; (void)n_in; (void)out_size; (void)ws_size;
  (void)w2b; (void)w3b;
}
